// myGConvLSTM_71923522339516
// MI455X (gfx1250) — matrix operand order repaired and checked by reading, not compiled, not run on hardware
//
#include <hip/hip_runtime.h>
#include <math.h>

// ---------------- problem constants (match reference) ----------------
#define KCH 3
#define BB  2
#define TT  12
#define NN  8192
#define FIN 8
#define HID 16
#define MX  (TT * BB * FIN)   // 192 columns for the x-stream
#define MH  (BB * HID)        // 32 columns for the h-stream

// GEMM tiling
#define KC     256            // K elements per TDM chunk
#define KCP    (KC + 8)       // padded LDS row (264 elem = 132 dwords; 132%64==4 -> bank spread)
#define GWAVES 4              // waves per GEMM block

typedef __attribute__((ext_vector_type(16))) __bf16 v16bf;
typedef __attribute__((ext_vector_type(8)))  __bf16 v8bf;
typedef __attribute__((ext_vector_type(4)))  __bf16 v4bf;
typedef __attribute__((ext_vector_type(8)))  float  v8f;
typedef __attribute__((ext_vector_type(4)))  unsigned int u32x4;
typedef __attribute__((ext_vector_type(8)))  int i32x8;
typedef __attribute__((ext_vector_type(4)))  int i32x4;

// ---------------------------------------------------------------------
// TDM: async-load a [32 x KCP] bf16 panel (column stride NN elements)
// from global into LDS.  Descriptor per CDNA5 ISA ch.8 (D# groups 0/1).
// ---------------------------------------------------------------------
__device__ __forceinline__ void tdm_load_panel(const __bf16* gsrc, unsigned lds_byte_off) {
    unsigned long long ga = (unsigned long long)(uintptr_t)gsrc;
    u32x4 g0;
    g0[0] = 1u;                                            // count=1 (valid, user mode)
    g0[1] = lds_byte_off;                                  // lds_addr (bytes)
    g0[2] = (unsigned)ga;                                  // global_addr[31:0]
    g0[3] = (unsigned)((ga >> 32) & 0x1FFFFFFull)          // global_addr[56:32]
          | (2u << 30);                                    // type = 2 ("image")
    i32x8 g1;
    g1[0] = (1 << 16);          // workgroup_mask=0, data_size=1 (2 bytes)
    g1[1] = (KCP << 16);        // tensor_dim0[15:0] = KCP
    g1[2] = (32  << 16);        // tensor_dim0[31:16]=0, tensor_dim1[15:0]=32
    g1[3] = (KCP << 16);        // tensor_dim1[31:16]=0, tile_dim0=KCP
    g1[4] = 32;                 // tile_dim1=32, tile_dim2=0
    g1[5] = NN;                 // tensor_dim0_stride[31:0] = NN (elements)
    g1[6] = 0;                  // stride0[47:32]=0, stride1[15:0]=0
    g1[7] = 0;                  // stride1[47:16]=0
    i32x4 gz = {0, 0, 0, 0};
#if __clang_major__ >= 23
    i32x8 gz8 = {0, 0, 0, 0, 0, 0, 0, 0};
    __builtin_amdgcn_tensor_load_to_lds(g0, g1, gz, gz, gz8, 0);
#else
    __builtin_amdgcn_tensor_load_to_lds(g0, g1, gz, gz, 0);
#endif
}

// ---------------------------------------------------------------------
// L fp32 -> bf16 (128 MB, L2-resident afterwards)
// ---------------------------------------------------------------------
__global__ __launch_bounds__(256)
void gcl_cvt_l(const float4* __restrict__ src, v4bf* __restrict__ dst, int n4) {
    int i = blockIdx.x * 256 + threadIdx.x;
    if (i < n4) {
        float4 v = src[i];
        v4bf o;
        o[0] = (__bf16)v.x; o[1] = (__bf16)v.y;
        o[2] = (__bf16)v.z; o[3] = (__bf16)v.w;
        dst[i] = o;
    }
}

// ---------------------------------------------------------------------
// X [B,T,N,FIN] f32 -> Xc [MX, N] bf16, col = (t*B+b)*FIN+f
// ---------------------------------------------------------------------
__global__ __launch_bounds__(256)
void gcl_build_xcols(const float* __restrict__ X, __bf16* __restrict__ Xc) {
    int idx = blockIdx.x * 256 + threadIdx.x;          // < MX*NN
    int col = idx >> 13;                                // / NN
    int n   = idx & (NN - 1);
    int t   = col / (BB * FIN);
    int r   = col % (BB * FIN);
    int b   = r / FIN;
    int f   = r % FIN;
    Xc[idx] = (__bf16)X[((((size_t)b * TT) + t) * NN + n) * FIN + f];
}

// ---------------------------------------------------------------------
// h [B,N,HID] f32 -> Hc [MH, N] bf16, col = b*HID + j
// ---------------------------------------------------------------------
__global__ __launch_bounds__(256)
void gcl_build_hcols(const float* __restrict__ h, __bf16* __restrict__ Hc) {
    int idx = blockIdx.x * 256 + threadIdx.x;          // < MH*NN
    int col = idx >> 13;
    int n   = idx & (NN - 1);
    int b   = col / HID;
    int j   = col % HID;
    Hc[idx] = (__bf16)h[(((size_t)b * NN) + n) * HID + j];
}

// ---------------------------------------------------------------------
// Y[col][n] = alpha * sum_m L[n,m] * V[col][m]  -  U[col][n]   (U optional)
// bf16 WMMA, fp32 accumulate.  Each wave: 16x32 tile (2 accumulators).
// B panel (32 cols x KC chunk) staged in LDS via TDM, double-buffered.
// grid = (NN/16/GWAVES, cols/32), block = GWAVES*32
// ---------------------------------------------------------------------
__global__ __launch_bounds__(GWAVES * 32)
void gcl_gemm_LV(const __bf16* __restrict__ Lb,
                 const __bf16* __restrict__ V,
                 const __bf16* __restrict__ U,
                 __bf16* __restrict__ Y,
                 float alpha) {
    __shared__ __bf16 Bsh[2][32 * KCP];

    const int lane  = threadIdx.x & 31;
    const int wave  = threadIdx.x >> 5;
    const int ntile = blockIdx.x * GWAVES + wave;      // 0..NN/16-1
    const int c0    = blockIdx.y * 32;                 // first column of panel
    const int row   = ntile * 16 + (lane & 15);        // A row (n)
    const int hi    = lane >> 4;                       // 0 / 1
    const int asel  = hi * 16;                         // A: K 0..15 / 16..31
    const int bsel  = hi * 16;                         // B: K 0..15 / 16..31

    const __bf16* aptr = Lb + (size_t)row * NN;
    const __bf16* gB   = V  + (size_t)c0  * NN;
    const unsigned lds0 = (unsigned)(uintptr_t)(&Bsh[0][0]);
    const unsigned lds1 = (unsigned)(uintptr_t)(&Bsh[1][0]);

    const int NCHUNK = NN / KC;                        // 32

    if (wave == 0)
        tdm_load_panel(gB, lds0);                      // chunk 0 -> buf 0

    v8f acc0 = {};
    v8f acc1 = {};

    for (int i = 0; i < NCHUNK; ++i) {
        __syncthreads();                               // buf[(i+1)&1] fully consumed
        if (wave == 0) {
            if (i + 1 < NCHUNK) {
                tdm_load_panel(gB + (size_t)(i + 1) * KC, ((i + 1) & 1) ? lds1 : lds0);
                __builtin_amdgcn_s_wait_tensorcnt(1);  // chunk i complete (in-order)
            } else {
                __builtin_amdgcn_s_wait_tensorcnt(0);
            }
        }
        __syncthreads();                               // chunk i visible in buf[i&1]

        const __bf16* Bc = &Bsh[i & 1][0];
        const __bf16* ac = aptr + i * KC + asel;
        const __bf16* bl0 = Bc + ((lane & 15)      ) * KCP + bsel;
        const __bf16* bl1 = Bc + ((lane & 15) + 16 ) * KCP + bsel;

        #pragma unroll
        for (int kb = 0; kb < KC; kb += 32) {
            v16bf a, b0, b1;
            *((v8bf*)&a  + 0) = *(const v8bf*)(ac  + kb);
            *((v8bf*)&a  + 1) = *(const v8bf*)(ac  + kb + 8);
            *((v8bf*)&b0 + 0) = *(const v8bf*)(bl0 + kb);
            *((v8bf*)&b0 + 1) = *(const v8bf*)(bl0 + kb + 8);
            *((v8bf*)&b1 + 0) = *(const v8bf*)(bl1 + kb);
            *((v8bf*)&b1 + 1) = *(const v8bf*)(bl1 + kb + 8);
            acc0 = __builtin_amdgcn_wmma_f32_16x16x32_bf16(
                       false, a, false, b0, (short)0, acc0, false, false);
            acc1 = __builtin_amdgcn_wmma_f32_16x16x32_bf16(
                       false, a, false, b1, (short)0, acc1, false, false);
        }
    }

    // D layout: VGPR j -> row = ntile*16 + j + 8*hi, col = (lane&15) of each tile
    const int rbase = ntile * 16 + hi * 8;
    #pragma unroll
    for (int ctile = 0; ctile < 2; ++ctile) {
        const v8f acc = ctile ? acc1 : acc0;
        const int col = c0 + ctile * 16 + (lane & 15);
        __bf16* yp = Y + (size_t)col * NN + rbase;
        v8bf outv;
        if (U) {
            const __bf16* up = U + (size_t)col * NN + rbase;
            #pragma unroll
            for (int j = 0; j < 8; ++j)
                outv[j] = (__bf16)(alpha * acc[j] - (float)up[j]);
        } else {
            #pragma unroll
            for (int j = 0; j < 8; ++j)
                outv[j] = (__bf16)(alpha * acc[j]);
        }
        *(v8bf*)yp = outv;
    }
}

// ---------------------------------------------------------------------
// Per-timestep LSTM cell: gate contraction (tiny, VALU) + pointwise update.
// One thread per (b,n) row.  Weight pointers are pre-offset to timestep t.
// ---------------------------------------------------------------------
__global__ __launch_bounds__(128)
void gcl_cell(const __bf16* __restrict__ Xc0, const __bf16* __restrict__ Xc1,
              const __bf16* __restrict__ Xc2,
              const __bf16* __restrict__ Hc0, const __bf16* __restrict__ Hc1,
              const __bf16* __restrict__ Hc2,
              const float* __restrict__ cprev,      // [B,N,HID]
              const float* __restrict__ Wx,         // [4,K,FIN,HID]
              const float* __restrict__ Wh,         // [4,K,HID,HID]
              const float* __restrict__ bx, const float* __restrict__ bh,
              const float* __restrict__ wc, const float* __restrict__ bg,
              float* __restrict__ out,              // d_out base [B,T,N,HID]
              float* __restrict__ hs_t,             // hs + t*B*N*HID
              float* __restrict__ cs_t,             // cs + t*B*N*HID
              int t) {
    int idx = blockIdx.x * 128 + threadIdx.x;       // < B*N
    if (idx >= BB * NN) return;
    int b = idx / NN;
    int n = idx % NN;

    float acc[4][HID];
    #pragma unroll
    for (int g = 0; g < 4; ++g)
        #pragma unroll
        for (int j = 0; j < HID; ++j)
            acc[g][j] = bx[g * HID + j] + bh[g * HID + j] + bg[g * HID + j];

    const __bf16* Xcs[KCH] = {Xc0, Xc1, Xc2};
    const __bf16* Hcs[KCH] = {Hc0, Hc1, Hc2};

    for (int k = 0; k < KCH; ++k) {
        #pragma unroll
        for (int f = 0; f < FIN; ++f) {
            float xv = (float)Xcs[k][((size_t)((t * BB + b) * FIN + f)) * NN + n];
            #pragma unroll
            for (int g = 0; g < 4; ++g) {
                const float* wg = Wx + ((g * KCH + k) * FIN + f) * HID;
                #pragma unroll
                for (int j = 0; j < HID; ++j)
                    acc[g][j] = fmaf(xv, wg[j], acc[g][j]);
            }
        }
        #pragma unroll
        for (int i = 0; i < HID; ++i) {
            float hv = (float)Hcs[k][((size_t)(b * HID + i)) * NN + n];
            #pragma unroll
            for (int g = 0; g < 4; ++g) {
                const float* wg = Wh + ((g * KCH + k) * HID + i) * HID;
                #pragma unroll
                for (int j = 0; j < HID; ++j)
                    acc[g][j] = fmaf(hv, wg[j], acc[g][j]);
            }
        }
    }

    size_t rowoff = (size_t)(b * NN + n) * HID;
    size_t outoff = (size_t)((b * TT + t) * NN + n) * HID;
    #pragma unroll
    for (int j = 0; j < HID; ++j) {
        float c0 = cprev[rowoff + j];
        float ig = 1.0f / (1.0f + expf(-(acc[0][j] + wc[0 * HID + j] * c0)));
        float fg = 1.0f / (1.0f + expf(-(acc[1][j] + wc[1 * HID + j] * c0)));
        float cn = fg * c0 + ig * tanhf(acc[2][j]);
        float og = 1.0f / (1.0f + expf(-(acc[3][j] + wc[2 * HID + j] * cn)));
        float hn = og * tanhf(cn);
        hs_t[rowoff + j] = hn;
        cs_t[rowoff + j] = cn;
        out[outoff + j]  = hn;
    }
}

// ---------------------------------------------------------------------
extern "C" void kernel_launch(void* const* d_in, const int* in_sizes, int n_in,
                              void* d_out, int out_size, void* d_ws, size_t ws_size,
                              hipStream_t stream) {
    const float* X  = (const float*)d_in[0];   // [B,T,N,FIN]
    const float* L  = (const float*)d_in[1];   // [N,N]
    const float* H  = (const float*)d_in[2];   // [T,B,N,HID] (zeros; H[0] used)
    const float* C  = (const float*)d_in[3];   // [T,B,N,HID]
    const float* Wx = (const float*)d_in[4];   // [T,4,K,FIN,HID]
    const float* Wh = (const float*)d_in[5];   // [T,4,K,HID,HID]
    const float* bx = (const float*)d_in[6];   // [T,4,HID]
    const float* bh = (const float*)d_in[7];   // [T,4,HID]
    const float* wc = (const float*)d_in[8];   // [T,3,HID]
    const float* bg = (const float*)d_in[9];   // [T,4,HID]

    float* out = (float*)d_out;                          // [B,T,N,HID]
    float* hs  = out + (size_t)BB * TT * NN * HID;       // [T,B,N,HID]
    float* cs  = hs  + (size_t)TT * BB * NN * HID;       // [T,B,N,HID]

    // workspace carve-up (bf16 elements)
    __bf16* p   = (__bf16*)d_ws;
    __bf16* Lb  = p; p += (size_t)NN * NN;   // 128 MB
    __bf16* Xc0 = p; p += (size_t)MX * NN;   // 3 MB each
    __bf16* Xc1 = p; p += (size_t)MX * NN;
    __bf16* Xc2 = p; p += (size_t)MX * NN;
    __bf16* Hc0 = p; p += (size_t)MH * NN;   // 0.5 MB each
    __bf16* Hc1 = p; p += (size_t)MH * NN;
    __bf16* Hc2 = p; p += (size_t)MH * NN;

    // 1) L -> bf16
    {
        int n4 = (NN * NN) / 4;                              // 16,777,216
        gcl_cvt_l<<<n4 / 256, 256, 0, stream>>>((const float4*)L, (v4bf*)Lb, n4);
    }
    // 2) X -> columns (T0x)
    gcl_build_xcols<<<(MX * NN) / 256, 256, 0, stream>>>(X, Xc0);

    // 3) x-stream Chebyshev (all timesteps at once)
    dim3 gx(NN / 16 / GWAVES, MX / 32);                       // (128, 6)
    gcl_gemm_LV<<<gx, GWAVES * 32, 0, stream>>>(Lb, Xc0, nullptr, Xc1, 1.0f);
    gcl_gemm_LV<<<gx, GWAVES * 32, 0, stream>>>(Lb, Xc1, Xc0,     Xc2, 2.0f);

    // 4) recurrent loop
    dim3 gh(NN / 16 / GWAVES, MH / 32);                       // (128, 1)
    for (int t = 0; t < TT; ++t) {
        const float* hprev = t ? (hs + (size_t)(t - 1) * BB * NN * HID) : H;
        const float* cprev = t ? (cs + (size_t)(t - 1) * BB * NN * HID) : C;

        gcl_build_hcols<<<(MH * NN) / 256, 256, 0, stream>>>(hprev, Hc0);
        gcl_gemm_LV<<<gh, GWAVES * 32, 0, stream>>>(Lb, Hc0, nullptr, Hc1, 1.0f);
        gcl_gemm_LV<<<gh, GWAVES * 32, 0, stream>>>(Lb, Hc1, Hc0,     Hc2, 2.0f);

        gcl_cell<<<(BB * NN) / 128, 128, 0, stream>>>(
            Xc0, Xc1, Xc2, Hc0, Hc1, Hc2, cprev,
            Wx + (size_t)t * 4 * KCH * FIN * HID,
            Wh + (size_t)t * 4 * KCH * HID * HID,
            bx + (size_t)t * 4 * HID,
            bh + (size_t)t * 4 * HID,
            wc + (size_t)t * 3 * HID,
            bg + (size_t)t * 4 * HID,
            out,
            hs + (size_t)t * BB * NN * HID,
            cs + (size_t)t * BB * NN * HID,
            t);
    }
}
